// ResnetBlocWithAttn_89060441850327
// MI455X (gfx1250) — compile-verified, matrix operand order checked by reading, not run on hardware
//
#include <hip/hip_runtime.h>
#include <math.h>

// ---------------------------------------------------------------------------
// Problem constants: B=16, C=256, H=W=32, GROUPS=32, NE=256
// ---------------------------------------------------------------------------
#define BATCH   16
#define CCH     256
#define HW      1024
#define NTOT    (BATCH * HW)
#define GRPSZ   (8 * HW)      // elements per (b,group)
#define LDK     40            // ushorts per LDS row: 80B -> 16B-aligned rows

typedef __attribute__((ext_vector_type(16))) __bf16 v16bf;
typedef __attribute__((ext_vector_type(8)))  float  v8f;
typedef __attribute__((ext_vector_type(4)))  int    v4i;
typedef __attribute__((address_space(3)))    v4i    as3_v4i;

// ---------------------------------------------------------------------------
// Async global->LDS staging (CDNA5). Falls back to sync uint4 copies if the
// builtin is unavailable on this toolchain.
// Probe-derived signature: (v4i* gsrc, v4i addrspace(3)* ldst, imm, imm cpol)
// ---------------------------------------------------------------------------
#if defined(__has_builtin)
#if __has_builtin(__builtin_amdgcn_global_load_async_to_lds_b128)
#define HAVE_ASYNC_LDS 1
#endif
#endif
#ifndef HAVE_ASYNC_LDS
#define HAVE_ASYNC_LDS 0
#endif

#if HAVE_ASYNC_LDS
__device__ __forceinline__ void async_b128(const void* g, void* l) {
  __builtin_amdgcn_global_load_async_to_lds_b128((v4i*)g, (as3_v4i*)l, 0, 0);
}
__device__ __forceinline__ void wait_async() {
#if __has_builtin(__builtin_amdgcn_s_wait_asynccnt)
  __builtin_amdgcn_s_wait_asynccnt(0);
#else
  asm volatile("s_wait_asynccnt 0" ::: "memory");
#endif
}
#else
__device__ __forceinline__ void async_b128(const void* g, void* l) {
  *(uint4*)l = *(const uint4*)g;
}
__device__ __forceinline__ void wait_async() {}
#endif

// ---------------------------------------------------------------------------
// Helpers
// ---------------------------------------------------------------------------
__device__ __forceinline__ unsigned short f32_to_bf16(float f) {
  unsigned int u = __float_as_uint(f);
  unsigned int r = u + 0x7FFFu + ((u >> 16) & 1u);
  return (unsigned short)(r >> 16);
}

__device__ __forceinline__ v8f wmma_bf16(v16bf a, v16bf b, v8f c) {
  return __builtin_amdgcn_wmma_f32_16x16x32_bf16(
      false, a, false, b, (short)0, c, false, false);
}

// 16x32 bf16 fragment from row-major LDS tile [rows][LDK]; 2x ds_load_b128.
__device__ __forceinline__ v16bf load_frag(const unsigned short* lds, int row0) {
  int lane = threadIdx.x & 31;
  const unsigned short* p = lds + (row0 + (lane & 15)) * LDK + ((lane >> 4) << 3);
  union { v16bf v; uint4 q[2]; } f;
  f.q[0] = *(const uint4*)(p);
  f.q[1] = *(const uint4*)(p + 16);
  return f.v;
}

// Conv3x3 B fragment straight out of the zero-padded halo tile sH[ry][xp][k].
__device__ __forceinline__ v16bf load_frag_halo(const unsigned short* sH,
                                                int nt, int dy, int dx) {
  int lane = threadIdx.x & 31;
  int n  = nt + (lane & 15);
  int ry = (n >> 5) + dy + 1;
  int xp = (n & 31) + dx + 1;
  const unsigned short* p = sH + (ry * 34 + xp) * LDK + ((lane >> 4) << 3);
  union { v16bf v; uint4 q[2]; } f;
  f.q[0] = *(const uint4*)(p);
  f.q[1] = *(const uint4*)(p + 16);
  return f.v;
}

// 64x64 core (conv3x3): wave w -> m-tile (w&3), n-tiles {2*(w>>2), +1}
__device__ __forceinline__ void mma_step(const unsigned short* sA,
                                         const unsigned short* sB,
                                         int wid, v8f acc[2]) {
  v16bf a  = load_frag(sA, (wid & 3) * 16);
  v16bf b0 = load_frag(sB, ((wid >> 2) * 2 + 0) * 16);
  v16bf b1 = load_frag(sB, ((wid >> 2) * 2 + 1) * 16);
  acc[0] = wmma_bf16(a, b0, acc[0]);
  acc[1] = wmma_bf16(a, b1, acc[1]);
}

// 128x64 core: wave w owns m-tile w (16 rows) x all 4 n-tiles.
__device__ __forceinline__ void mma_step8(const unsigned short* sA,
                                          const unsigned short* sB,
                                          int wid, v8f acc[4]) {
  v16bf a = load_frag(sA, wid * 16);
#pragma unroll
  for (int j = 0; j < 4; ++j) {
    v16bf bj = load_frag(sB, j * 16);
    acc[j] = wmma_bf16(a, bj, acc[j]);
  }
}

// ---------------------------------------------------------------------------
// Staging helpers
// ---------------------------------------------------------------------------
template <int ROWS>
__device__ __forceinline__ void stage_k(unsigned short* sD, const unsigned short* src,
                                        size_t rstride, int tid) {
  for (int t = tid; t < ROWS * 4; t += 256) {
    int row = t >> 2, colg = (t & 3) * 8;
    async_b128(src + (size_t)row * rstride + colg, sD + row * LDK + colg);
  }
}

template <int ROWS>
__device__ __forceinline__ void stage_kT(unsigned short* sD, const unsigned short* src,
                                         size_t kstride, int tid) {
  for (int t = tid; t < ROWS * 4; t += 256) {
    int kl  = t / (ROWS / 8);
    int nlg = (t % (ROWS / 8)) * 8;
    union { uint4 q; unsigned short e[8]; } v;
    v.q = *(const uint4*)(src + kstride * kl + nlg);
#pragma unroll
    for (int j = 0; j < 8; ++j) sD[(nlg + j) * LDK + kl] = v.e[j];
  }
}

// ---------------------------------------------------------------------------
// Weight conversion / small kernels
// ---------------------------------------------------------------------------
__global__ void k_cvt_bf16(const float* __restrict__ s, unsigned short* __restrict__ d, int n) {
  int i = (blockIdx.x * 256 + threadIdx.x) * 4;
  if (i < n) {
    float4 v = *(const float4*)(s + i);
    union { uint2 q; unsigned short e[4]; } o;
    o.e[0] = f32_to_bf16(v.x); o.e[1] = f32_to_bf16(v.y);
    o.e[2] = f32_to_bf16(v.z); o.e[3] = f32_to_bf16(v.w);
    *(uint2*)(d + i) = o.q;
  }
}

__global__ void k_cvt_conv3x3(const float* __restrict__ s, unsigned short* __restrict__ d) {
  int i = blockIdx.x * 256 + threadIdx.x;     // [tap][cout][cin] <- OIHW
  int ci = i & 255, co = (i >> 8) & 255, tap = i >> 16;
  d[i] = f32_to_bf16(s[(co * 256 + ci) * 9 + tap]);
}

__global__ void k_noise(const float* __restrict__ te, const float* __restrict__ w,
                        const float* __restrict__ nb, float* __restrict__ noise) {
  int i = blockIdx.x * 256 + threadIdx.x;
  int b = i >> 8, c = i & 255;
  float s = nb[c];
  for (int k = 0; k < 256; k += 4) {
    float4 a = *(const float4*)(te + b * 256 + k);
    float4 ww = *(const float4*)(w + c * 256 + k);
    s += a.x * ww.x + a.y * ww.y + a.z * ww.z + a.w * ww.w;
  }
  noise[i] = s;
}

// ---------------------------------------------------------------------------
// GroupNorm stats + fused normalize(+swish) -> bf16
// ---------------------------------------------------------------------------
__global__ void k_gnstats(const float* __restrict__ x, float* __restrict__ stats) {
  __shared__ float rs[8], rs2[8];
  int bg = blockIdx.x;
  const float* p = x + (size_t)bg * GRPSZ;
  float s = 0.f, s2 = 0.f;
  for (int i = threadIdx.x; i < GRPSZ / 4; i += 256) {
    float4 v = *(const float4*)(p + i * 4);
    s  += v.x + v.y + v.z + v.w;
    s2 += v.x * v.x + v.y * v.y + v.z * v.z + v.w * v.w;
  }
#pragma unroll
  for (int o = 16; o > 0; o >>= 1) { s += __shfl_xor(s, o); s2 += __shfl_xor(s2, o); }
  if ((threadIdx.x & 31) == 0) { rs[threadIdx.x >> 5] = s; rs2[threadIdx.x >> 5] = s2; }
  __syncthreads();
  if (threadIdx.x == 0) {
    float S = 0.f, S2 = 0.f;
    for (int i = 0; i < 8; ++i) { S += rs[i]; S2 += rs2[i]; }
    float m   = S * (1.f / GRPSZ);
    float var = S2 * (1.f / GRPSZ) - m * m;
    stats[bg]       = m;
    stats[512 + bg] = rsqrtf(var + 1e-5f);
  }
}

__global__ void k_gnact(const float* __restrict__ x, const float* __restrict__ stats,
                        const float* __restrict__ g, const float* __restrict__ bta,
                        unsigned short* __restrict__ out, int do_swish) {
  size_t base = ((size_t)blockIdx.x * 256 + threadIdx.x) * 4;
  int c  = (int)((base >> 10) & 255);
  int bg = (int)(base >> 13);
  float mu = stats[bg], rs = stats[512 + bg];
  float ga = g[c], be = bta[c];
  float4 v = *(const float4*)(x + base);
  float e0 = (v.x - mu) * rs * ga + be;
  float e1 = (v.y - mu) * rs * ga + be;
  float e2 = (v.z - mu) * rs * ga + be;
  float e3 = (v.w - mu) * rs * ga + be;
  if (do_swish) {
    e0 = e0 / (1.f + __expf(-e0));
    e1 = e1 / (1.f + __expf(-e1));
    e2 = e2 / (1.f + __expf(-e2));
    e3 = e3 / (1.f + __expf(-e3));
  }
  union { uint2 q; unsigned short e[4]; } o;
  o.e[0] = f32_to_bf16(e0); o.e[1] = f32_to_bf16(e1);
  o.e[2] = f32_to_bf16(e2); o.e[3] = f32_to_bf16(e3);
  *(uint2*)(out + base) = o.q;
}

// ---------------------------------------------------------------------------
// Conv3x3 implicit GEMM: 64 cout x 64 px block, K-chunk outer, 9 taps from
// an LDS halo; weights staged with async b128 copies overlapping the gather.
// ---------------------------------------------------------------------------
__global__ void k_conv3x3(const unsigned short* __restrict__ act,
                          const unsigned short* __restrict__ wb,   // [9][C][C]
                          const float* __restrict__ bias,
                          const float* __restrict__ noise,         // [B][C] or null
                          float* __restrict__ out,
                          int accumulate) {
  __shared__ __align__(16) unsigned short sA[9 * 64 * LDK];
  __shared__ __align__(16) unsigned short sH[4 * 34 * LDK];
  int mb = blockIdx.y, nb = blockIdx.x;
  int tid = threadIdx.x, wid = tid >> 5;
  int n0 = nb * 64;
  int b = n0 >> 10, hw0 = n0 & 1023, y0 = hw0 >> 5;
  v8f acc[2] = {{}, {}};

  for (int kc = 0; kc < CCH; kc += 32) {
    {   // async-stage 9 weight chunks (overlaps with halo gather below)
      int row = tid >> 2, colg = (tid & 3) * 8;
#pragma unroll
      for (int tap = 0; tap < 9; ++tap)
        async_b128(wb + (size_t)(tap * CCH + mb * 64 + row) * CCH + kc + colg,
                   sA + (tap * 64 + row) * LDK + colg);
    }
    for (int task = tid; task < 544; task += 256) {   // zero-padded halo
      int xp = task % 34;
      int t2 = task / 34;
      int kg = t2 & 3, ry = t2 >> 2;
      int y = y0 - 1 + ry, x = xp - 1;
      union { uint4 q; unsigned short e[8]; } t;
      if ((unsigned)y < 32u && (unsigned)x < 32u) {
        const unsigned short* src = act + ((size_t)(b * CCH + kc + kg * 8) << 10) + (y << 5) + x;
#pragma unroll
        for (int j = 0; j < 8; ++j) t.e[j] = src[(size_t)j << 10];
      } else {
        t.q = make_uint4(0u, 0u, 0u, 0u);
      }
      *(uint4*)(sH + (ry * 34 + xp) * LDK + kg * 8) = t.q;
    }
    wait_async();
    __syncthreads();
#pragma unroll
    for (int tap = 0; tap < 9; ++tap) {
      int dy = tap / 3 - 1, dx = tap % 3 - 1;
      v16bf a  = load_frag(sA + tap * 64 * LDK, (wid & 3) * 16);
      v16bf b0 = load_frag_halo(sH, ((wid >> 2) * 2 + 0) * 16, dy, dx);
      v16bf b1 = load_frag_halo(sH, ((wid >> 2) * 2 + 1) * 16, dy, dx);
      acc[0] = wmma_bf16(a, b0, acc[0]);
      acc[1] = wmma_bf16(a, b1, acc[1]);
    }
    __syncthreads();
  }
  int lane = tid & 31, half = lane >> 4, nloc = lane & 15;
#pragma unroll
  for (int j = 0; j < 2; ++j) {
    int gn = n0 + ((wid >> 2) * 2 + j) * 16 + nloc;
#pragma unroll
    for (int r = 0; r < 8; ++r) {
      int gm = mb * 64 + (wid & 3) * 16 + half * 8 + r;
      float v = acc[j][r] + bias[gm];
      if (noise) v += noise[b * CCH + gm];
      size_t o = ((size_t)(b * CCH + gm) << 10) + (gn & 1023);
      if (accumulate) out[o] += v; else out[o] = v;
    }
  }
}

// ---------------------------------------------------------------------------
// Flat GEMMs (128x64 block tiles)
// ---------------------------------------------------------------------------

// cfunc 1x1: HS = W*c + bias + x   (B converted f32->bf16 in flight)
__global__ void k_conv1x1_cf(const float* __restrict__ cin,
                             const unsigned short* __restrict__ w,
                             const float* __restrict__ bias,
                             const float* __restrict__ xres,
                             float* __restrict__ out) {
  __shared__ __align__(16) unsigned short sA[128 * LDK];
  __shared__ __align__(16) unsigned short sB[64 * LDK];
  int mb = blockIdx.y, nb = blockIdx.x;
  int tid = threadIdx.x, wid = tid >> 5;
  int m0 = mb * 128, n0 = nb * 64, b = n0 >> 10, hw0 = n0 & 1023;
  v8f acc[4] = {{}, {}, {}, {}};
  for (int kc = 0; kc < CCH; kc += 32) {
    stage_k<128>(sA, w + (size_t)m0 * CCH + kc, CCH, tid);
    {
      int nlg = (tid & 7) * 8, kl = tid >> 3;
      const float* src = cin + ((size_t)(b * CCH + kc + kl) << 10) + hw0 + nlg;
      float4 f0 = *(const float4*)(src);
      float4 f1 = *(const float4*)(src + 4);
      unsigned short e[8];
      e[0] = f32_to_bf16(f0.x); e[1] = f32_to_bf16(f0.y);
      e[2] = f32_to_bf16(f0.z); e[3] = f32_to_bf16(f0.w);
      e[4] = f32_to_bf16(f1.x); e[5] = f32_to_bf16(f1.y);
      e[6] = f32_to_bf16(f1.z); e[7] = f32_to_bf16(f1.w);
#pragma unroll
      for (int j = 0; j < 8; ++j) sB[(nlg + j) * LDK + kl] = e[j];
    }
    wait_async();
    __syncthreads();
    mma_step8(sA, sB, wid, acc);
    __syncthreads();
  }
  int lane = tid & 31, half = lane >> 4, nloc = lane & 15;
#pragma unroll
  for (int j = 0; j < 4; ++j) {
    int gn = n0 + j * 16 + nloc;
#pragma unroll
    for (int r = 0; r < 8; ++r) {
      int gm = m0 + wid * 16 + half * 8 + r;
      size_t o = ((size_t)(b * CCH + gm) << 10) + (gn & 1023);
      out[o] = acc[j][r] + bias[gm] + xres[o];
    }
  }
}

// qkv 1x1 (no bias), M=768 -> bf16
__global__ void k_qkv(const unsigned short* __restrict__ act,
                      const unsigned short* __restrict__ w,
                      unsigned short* __restrict__ qkv) {
  __shared__ __align__(16) unsigned short sA[128 * LDK];
  __shared__ __align__(16) unsigned short sB[64 * LDK];
  int mb = blockIdx.y, nb = blockIdx.x;
  int tid = threadIdx.x, wid = tid >> 5;
  int m0 = mb * 128, n0 = nb * 64, b = n0 >> 10, hw0 = n0 & 1023;
  v8f acc[4] = {{}, {}, {}, {}};
  for (int kc = 0; kc < CCH; kc += 32) {
    stage_k<128>(sA, w + (size_t)m0 * CCH + kc, CCH, tid);
    stage_kT<64>(sB, act + ((size_t)(b * CCH + kc) << 10) + hw0, (size_t)HW, tid);
    wait_async();
    __syncthreads();
    mma_step8(sA, sB, wid, acc);
    __syncthreads();
  }
  int lane = tid & 31, half = lane >> 4, nloc = lane & 15;
#pragma unroll
  for (int j = 0; j < 4; ++j) {
    int gn = n0 + j * 16 + nloc;
#pragma unroll
    for (int r = 0; r < 8; ++r) {
      int gm = m0 + wid * 16 + half * 8 + r;
      qkv[((size_t)b * 768 + gm) * HW + (gn & 1023)] = f32_to_bf16(acc[j][r]);
    }
  }
}

// scores[b][s][t] = q.k / 16 (per-batch 1024x1024, K=256)
__global__ void k_scores(const unsigned short* __restrict__ qkv,
                         float* __restrict__ scores) {
  __shared__ __align__(16) unsigned short sA[128 * LDK];
  __shared__ __align__(16) unsigned short sB[64 * LDK];
  int b = blockIdx.z, mb = blockIdx.y, nb = blockIdx.x;
  const unsigned short* q = qkv + (size_t)b * 768 * HW;
  const unsigned short* k = q + (size_t)CCH * HW;
  int tid = threadIdx.x, wid = tid >> 5;
  int m0 = mb * 128, n0 = nb * 64;
  v8f acc[4] = {{}, {}, {}, {}};
  for (int kc = 0; kc < CCH; kc += 32) {
    stage_kT<128>(sA, q + (size_t)kc * HW + m0, (size_t)HW, tid);
    stage_kT<64>(sB, k + (size_t)kc * HW + n0, (size_t)HW, tid);
    __syncthreads();
    mma_step8(sA, sB, wid, acc);
    __syncthreads();
  }
  int lane = tid & 31, half = lane >> 4, nloc = lane & 15;
#pragma unroll
  for (int j = 0; j < 4; ++j) {
    int gn = n0 + j * 16 + nloc;
#pragma unroll
    for (int r = 0; r < 8; ++r) {
      int gm = m0 + wid * 16 + half * 8 + r;
      scores[((size_t)b << 20) + (size_t)gm * HW + gn] = acc[j][r] * 0.0625f;
    }
  }
}

// Row softmax -> bf16 probs
__global__ void k_softmax(const float* __restrict__ s, unsigned short* __restrict__ p) {
  __shared__ float red[8];
  int row = blockIdx.x;
  int tid = threadIdx.x;
  const float* sr = s + (size_t)row * HW;
  float4 v = *(const float4*)(sr + tid * 4);
  float m = fmaxf(fmaxf(v.x, v.y), fmaxf(v.z, v.w));
#pragma unroll
  for (int o = 16; o > 0; o >>= 1) m = fmaxf(m, __shfl_xor(m, o));
  if ((tid & 31) == 0) red[tid >> 5] = m;
  __syncthreads();
  m = red[0];
#pragma unroll
  for (int i = 1; i < 8; ++i) m = fmaxf(m, red[i]);
  __syncthreads();
  v.x = __expf(v.x - m); v.y = __expf(v.y - m);
  v.z = __expf(v.z - m); v.w = __expf(v.w - m);
  float sum = v.x + v.y + v.z + v.w;
#pragma unroll
  for (int o = 16; o > 0; o >>= 1) sum += __shfl_xor(sum, o);
  if ((tid & 31) == 0) red[tid >> 5] = sum;
  __syncthreads();
  float tot = 0.f;
#pragma unroll
  for (int i = 0; i < 8; ++i) tot += red[i];
  float inv = 1.f / tot;
  union { uint2 q; unsigned short e[4]; } o;
  o.e[0] = f32_to_bf16(v.x * inv); o.e[1] = f32_to_bf16(v.y * inv);
  o.e[2] = f32_to_bf16(v.z * inv); o.e[3] = f32_to_bf16(v.w * inv);
  *(uint2*)(p + (size_t)row * HW + tid * 4) = o.q;
}

// o[b][c][s] = sum_t v[c][t] probs[s][t] (K=1024, both operands K-contig)
__global__ void k_pv(const unsigned short* __restrict__ qkv,
                     const unsigned short* __restrict__ probs,
                     unsigned short* __restrict__ obuf) {
  __shared__ __align__(16) unsigned short sA[128 * LDK];
  __shared__ __align__(16) unsigned short sB[64 * LDK];
  int b = blockIdx.z, mb = blockIdx.y, nb = blockIdx.x;
  const unsigned short* v = qkv + (size_t)b * 768 * HW + (size_t)2 * CCH * HW;
  const unsigned short* pb = probs + ((size_t)b << 20);
  int tid = threadIdx.x, wid = tid >> 5;
  int m0 = mb * 128, n0 = nb * 64;
  v8f acc[4] = {{}, {}, {}, {}};
  for (int kc = 0; kc < HW; kc += 32) {
    stage_k<128>(sA, v + (size_t)m0 * HW + kc, HW, tid);
    stage_k<64>(sB, pb + (size_t)n0 * HW + kc, HW, tid);
    wait_async();
    __syncthreads();
    mma_step8(sA, sB, wid, acc);
    __syncthreads();
  }
  int lane = tid & 31, half = lane >> 4, nloc = lane & 15;
#pragma unroll
  for (int j = 0; j < 4; ++j) {
    int gn = n0 + j * 16 + nloc;
#pragma unroll
    for (int r = 0; r < 8; ++r) {
      int gm = m0 + wid * 16 + half * 8 + r;
      obuf[((size_t)(b * CCH + gm) << 10) + gn] = f32_to_bf16(acc[j][r]);
    }
  }
}

// out 1x1: d_out = W*o + bias + HS
__global__ void k_outconv(const unsigned short* __restrict__ act,
                          const unsigned short* __restrict__ w,
                          const float* __restrict__ bias,
                          const float* __restrict__ hs,
                          float* __restrict__ out) {
  __shared__ __align__(16) unsigned short sA[128 * LDK];
  __shared__ __align__(16) unsigned short sB[64 * LDK];
  int mb = blockIdx.y, nb = blockIdx.x;
  int tid = threadIdx.x, wid = tid >> 5;
  int m0 = mb * 128, n0 = nb * 64, b = n0 >> 10, hw0 = n0 & 1023;
  v8f acc[4] = {{}, {}, {}, {}};
  for (int kc = 0; kc < CCH; kc += 32) {
    stage_k<128>(sA, w + (size_t)m0 * CCH + kc, CCH, tid);
    stage_kT<64>(sB, act + ((size_t)(b * CCH + kc) << 10) + hw0, (size_t)HW, tid);
    wait_async();
    __syncthreads();
    mma_step8(sA, sB, wid, acc);
    __syncthreads();
  }
  int lane = tid & 31, half = lane >> 4, nloc = lane & 15;
#pragma unroll
  for (int j = 0; j < 4; ++j) {
    int gn = n0 + j * 16 + nloc;
#pragma unroll
    for (int r = 0; r < 8; ++r) {
      int gm = m0 + wid * 16 + half * 8 + r;
      size_t o = ((size_t)(b * CCH + gm) << 10) + (gn & 1023);
      out[o] = acc[j][r] + bias[gm] + hs[o];
    }
  }
}

// ---------------------------------------------------------------------------
// Host-side pipeline
// ---------------------------------------------------------------------------
extern "C" void kernel_launch(void* const* d_in, const int* in_sizes, int n_in,
                              void* d_out, int out_size, void* d_ws, size_t ws_size,
                              hipStream_t stream) {
  const float* x        = (const float*)d_in[0];
  const float* time_emb = (const float*)d_in[1];
  const float* cmap     = (const float*)d_in[2];
  const float* gn1_g    = (const float*)d_in[3];
  const float* gn1_b    = (const float*)d_in[4];
  const float* conv1_w  = (const float*)d_in[5];
  const float* conv1_b  = (const float*)d_in[6];
  const float* noise_w  = (const float*)d_in[7];
  const float* noise_b  = (const float*)d_in[8];
  const float* gn2_g    = (const float*)d_in[9];
  const float* gn2_b    = (const float*)d_in[10];
  const float* conv2_w  = (const float*)d_in[11];
  const float* conv2_b  = (const float*)d_in[12];
  const float* cfunc_w  = (const float*)d_in[13];
  const float* cfunc_b  = (const float*)d_in[14];
  const float* agn_g    = (const float*)d_in[15];
  const float* agn_b    = (const float*)d_in[16];
  const float* qkv_w    = (const float*)d_in[17];
  const float* out_w    = (const float*)d_in[18];
  const float* out_b    = (const float*)d_in[19];
  float* out = (float*)d_out;
  (void)in_sizes; (void)n_in; (void)out_size; (void)ws_size;

  char* ws = (char*)d_ws;
  size_t off = 0;
  auto take = [&](size_t bytes) -> char* {
    char* p = ws + off;
    off += (bytes + 255) & ~(size_t)255;
    return p;
  };
  float*          H1     = (float*)take((size_t)BATCH * CCH * HW * 4);
  float*          HS     = (float*)take((size_t)BATCH * CCH * HW * 4);
  unsigned short* ABUF   = (unsigned short*)take((size_t)BATCH * CCH * HW * 2);
  unsigned short* QKV    = (unsigned short*)take((size_t)BATCH * 3 * CCH * HW * 2);
  float*          SCORES = (float*)take((size_t)BATCH * HW * HW * 4);
  unsigned short* PROBS  = (unsigned short*)take((size_t)BATCH * HW * HW * 2);
  unsigned short* WB1    = (unsigned short*)take((size_t)9 * CCH * CCH * 2);
  unsigned short* WB2    = (unsigned short*)take((size_t)9 * CCH * CCH * 2);
  unsigned short* WCF    = (unsigned short*)take((size_t)CCH * CCH * 2);
  unsigned short* WQKVW  = (unsigned short*)take((size_t)3 * CCH * CCH * 2);
  unsigned short* WOUTW  = (unsigned short*)take((size_t)CCH * CCH * 2);
  float*          NOISE  = (float*)take((size_t)BATCH * CCH * 4);
  float*          STATS  = (float*)take(1024 * 4);

  // --- weight prep ---
  k_cvt_conv3x3<<<2304, 256, 0, stream>>>(conv1_w, WB1);
  k_cvt_conv3x3<<<2304, 256, 0, stream>>>(conv2_w, WB2);
  k_cvt_bf16<<<64, 256, 0, stream>>>(cfunc_w, WCF, CCH * CCH);
  k_cvt_bf16<<<192, 256, 0, stream>>>(qkv_w, WQKVW, 3 * CCH * CCH);
  k_cvt_bf16<<<64, 256, 0, stream>>>(out_w, WOUTW, CCH * CCH);
  k_noise<<<16, 256, 0, stream>>>(time_emb, noise_w, noise_b, NOISE);

  // --- ResnetBlock ---
  k_gnstats<<<512, 256, 0, stream>>>(x, STATS);
  k_gnact<<<4096, 256, 0, stream>>>(x, STATS, gn1_g, gn1_b, ABUF, 1);
  k_conv3x3<<<dim3(256, 4), 256, 0, stream>>>(ABUF, WB1, conv1_b, NOISE, H1, 0);

  k_gnstats<<<512, 256, 0, stream>>>(H1, STATS);
  k_gnact<<<4096, 256, 0, stream>>>(H1, STATS, gn2_g, gn2_b, ABUF, 1);
  k_conv1x1_cf<<<dim3(256, 2), 256, 0, stream>>>(cmap, WCF, cfunc_b, x, HS);
  k_conv3x3<<<dim3(256, 4), 256, 0, stream>>>(ABUF, WB2, conv2_b, nullptr, HS, 1);

  // --- SelfAttention ---
  k_gnstats<<<512, 256, 0, stream>>>(HS, STATS);
  k_gnact<<<4096, 256, 0, stream>>>(HS, STATS, agn_g, agn_b, ABUF, 0);
  k_qkv<<<dim3(256, 6), 256, 0, stream>>>(ABUF, WQKVW, QKV);
  k_scores<<<dim3(16, 8, 16), 256, 0, stream>>>(QKV, SCORES);
  k_softmax<<<NTOT, 256, 0, stream>>>(SCORES, PROBS);
  k_pv<<<dim3(16, 2, 16), 256, 0, stream>>>(QKV, PROBS, ABUF);
  k_outconv<<<dim3(256, 2), 256, 0, stream>>>(ABUF, WOUTW, out_b, HS, out);
}
